// dgphgcn1_sparse_32006096290319
// MI455X (gfx1250) — matrix operand order repaired and checked by reading, not compiled, not run on hardware
//
#include <hip/hip_runtime.h>
#include <hip/hip_bf16.h>
#include <stdint.h>

// ---------------- problem constants ----------------
#define N_   32
#define C_   256
#define T_   128
#define V_   25
#define K_   3
#define MID_ 64
#define KM_  192     // K_*MID_
#define OUT_ 256
#define SP_  3200    // T_*V_ spatial per sample
#define VP_  32      // padded V for WMMA K-dim

#define RS_BN 0.9999950000374997f  // 1/sqrt(1+1e-5)

typedef __attribute__((ext_vector_type(16))) __bf16 v16bf;
typedef __attribute__((ext_vector_type(8)))  __bf16 v8bf;
typedef __attribute__((ext_vector_type(8)))  float  v8f;
typedef __attribute__((ext_vector_type(4)))  unsigned int v4u;
typedef __attribute__((ext_vector_type(8)))  int v8i;
typedef __attribute__((ext_vector_type(4)))  int v4i;

// ---------------- TDM availability ----------------
#if defined(__has_builtin)
#  if __has_builtin(__builtin_amdgcn_tensor_load_to_lds) && __has_builtin(__builtin_amdgcn_s_wait_tensorcnt)
#    define HAVE_TDM 1
#  endif
#endif
#ifndef HAVE_TDM
#  define HAVE_TDM 0
#endif

// ---------------- workspace layout (bytes) ----------------
constexpr size_t SZ_XBF  = (size_t)N_*C_*T_*V_*2;      // x in bf16, [n][c][t][v]
constexpr size_t SZ_PREX = (size_t)N_*KM_*T_*VP_*2;    // pre_x bf16, [n][kc][t][VP]
constexpr size_t SZ_Y    = (size_t)KM_*N_*T_*V_*2;     // y bf16, [kc][n][t][v]
constexpr size_t SZ_BAGG = (size_t)N_*K_*MID_*VP_*VP_*2; // adjacency B-operand bf16
constexpr size_t SZ_TMP  = (size_t)N_*C_*V_*4;         // mean over T, f32
constexpr size_t SZ_X12  = (size_t)N_*KM_*V_*4;        // x1/x2 f32
constexpr size_t SZ_PW   = (size_t)KM_*C_*2;           // pre_w bf16
constexpr size_t SZ_QW   = (size_t)OUT_*KM_*2;         // post_w bf16

constexpr size_t OFF_XBF  = 0;
constexpr size_t OFF_PREX = OFF_XBF  + SZ_XBF;
constexpr size_t OFF_Y    = OFF_PREX + SZ_PREX;
constexpr size_t OFF_BAGG = OFF_Y    + SZ_Y;
constexpr size_t OFF_TMP  = OFF_BAGG + SZ_BAGG;
constexpr size_t OFF_X1   = OFF_TMP  + SZ_TMP;
constexpr size_t OFF_X2   = OFF_X1   + SZ_X12;
constexpr size_t OFF_PW   = OFF_X2   + SZ_X12;
constexpr size_t OFF_QW   = OFF_PW   + SZ_PW;

// ---------------- WMMA helpers ----------------
__device__ inline v8f vzero8() {
  v8f z;
#pragma unroll
  for (int i = 0; i < 8; ++i) z[i] = 0.0f;
  return z;
}

__device__ inline v8f wmma_bf16(v16bf a, v16bf b, v8f c) {
  // D = A(16x32 bf16) x B(32x16 bf16) + C(16x16 f32)
  return __builtin_amdgcn_wmma_f32_16x16x32_bf16(false, a, false, b, (short)0, c, false, false);
}

// A-frag (16x32 bf16) per ISA layout: lane holds row M=lane%16;
// halves j<8  -> K = j + (hi?8:0)
// halves j>=8 -> K = (j-8) + 16 + (hi?8:0)
__device__ inline v16bf load_afrag(const __bf16* rowptr, int hi) {
  const v8bf a0 = *(const v8bf*)(rowptr + (hi ? 8 : 0));
  const v8bf a1 = *(const v8bf*)(rowptr + 16 + (hi ? 8 : 0));
  v16bf r;
#pragma unroll
  for (int i = 0; i < 8; ++i) { r[i] = a0[i]; r[8 + i] = a1[i]; }
  return r;
}

// ---------------- TDM 2D tile -> LDS ----------------
#if HAVE_TDM
__device__ inline void tdm_load_2d(unsigned lds_off, const void* gptr,
                                   unsigned tile0, unsigned tile1,
                                   unsigned td0, unsigned td1, unsigned stride0) {
  unsigned long long ga = (unsigned long long)(uintptr_t)gptr;
  v4u g0;
  g0[0] = 1u;                                           // count=1, user desc
  g0[1] = lds_off;                                      // lds_addr
  g0[2] = (unsigned)ga;                                 // global_addr[31:0]
  g0[3] = (unsigned)((ga >> 32) & 0x1FFFFFFu) | (2u << 30); // addr[56:32] | type=2
  v8i g1;
  g1[0] = (int)(1u << 16);                              // data_size=1 (2 bytes)
  g1[1] = (int)((td0 & 0xFFFFu) << 16);                 // tensor_dim0[15:0]
  g1[2] = (int)(((td0 >> 16) & 0xFFFFu) | ((td1 & 0xFFFFu) << 16));
  g1[3] = (int)(((td1 >> 16) & 0xFFFFu) | ((tile0 & 0xFFFFu) << 16));
  g1[4] = (int)(tile1 & 0xFFFFu);                       // tile_dim1 (tile_dim2=0)
  g1[5] = (int)stride0;                                 // tensor_dim0_stride[31:0]
  g1[6] = 0;
  g1[7] = 0;
  v4i z4; z4[0] = z4[1] = z4[2] = z4[3] = 0;
  v8i z8;
#pragma unroll
  for (int i = 0; i < 8; ++i) z8[i] = 0;
  // 6-arg form (clang-23 / therock-10.0): groups 2,3 + extension group unused (2D tile)
  __builtin_amdgcn_tensor_load_to_lds(g0, g1, z4, z4, z8, 0);
}
#endif

// ================= K0: mean over T + x -> bf16 =================
__global__ __launch_bounds__(128) void k0_mean_cvt(const float* __restrict__ x,
                                                   __bf16* __restrict__ xbf,
                                                   float* __restrict__ tmp) {
  __shared__ float red[T_ * 26];
  const int nc = blockIdx.x;          // n*C_ + c
  const int t  = threadIdx.x;         // 0..127
  const float*  src = x   + (size_t)(nc * T_ + t) * V_;
  __bf16*       dst = xbf + (size_t)(nc * T_ + t) * V_;
#pragma unroll
  for (int v = 0; v < V_; ++v) {
    float f = src[v];
    dst[v] = (__bf16)f;
    red[t * 26 + v] = f;
  }
  __syncthreads();
  if (t < V_) {
    float s = 0.f;
    for (int i = 0; i < T_; ++i) s += red[i * 26 + t];
    tmp[(size_t)nc * V_ + t] = s * (1.0f / T_);
  }
}

// ================= Kw: weights -> bf16 =================
__global__ __launch_bounds__(256) void kw_cvt(const float* __restrict__ pw,
                                              const float* __restrict__ qw,
                                              __bf16* __restrict__ pwbf,
                                              __bf16* __restrict__ qwbf) {
  int i = blockIdx.x * 256 + threadIdx.x;  // 192 blocks -> 49152 = KM_*C_ = OUT_*KM_
  pwbf[i] = (__bf16)pw[i];
  qwbf[i] = (__bf16)qw[i];
}

// ================= K1: x1/x2 = conv1x1(mean) =================
__global__ __launch_bounds__(32) void k1_x1x2(const float* __restrict__ tmp,
                                              const float* __restrict__ c1w, const float* __restrict__ c1b,
                                              const float* __restrict__ c2w, const float* __restrict__ c2b,
                                              float* __restrict__ x1, float* __restrict__ x2) {
  const int b = blockIdx.x;           // n*KM_ + kc
  const int n = b / KM_, kc = b % KM_;
  const int v = threadIdx.x;
  if (v >= V_) return;
  float s1 = c1b[kc], s2 = c2b[kc];
  const float* tb = tmp + (size_t)n * C_ * V_ + v;
  const float* w1 = c1w + (size_t)kc * C_;
  const float* w2 = c2w + (size_t)kc * C_;
  for (int c = 0; c < C_; ++c) {
    float tv = tb[(size_t)c * V_];
    s1 = fmaf(w1[c], tv, s1);
    s2 = fmaf(w2[c], tv, s2);
  }
  x1[(size_t)b * V_ + v] = s1;
  x2[(size_t)b * V_ + v] = s2;
}

// ================= K2: build adjacency B-operand (bf16, v-row major) =========
// Bagg[n][k][c][vr][uc] = Afull[uc][vr] ; rows vr>=V_ zeroed so padded K is safe
__global__ __launch_bounds__(256) void k2_bagg(const float* __restrict__ x1,
                                               const float* __restrict__ x2,
                                               const float* __restrict__ Am,
                                               const float* __restrict__ alpha,
                                               const float* __restrict__ beta,
                                               __bf16* __restrict__ bagg) {
  __shared__ float sx1[MID_ * V_], sx2[MID_ * V_], gls[V_ * 26];
  const int b = blockIdx.x;           // n*K_ + k
  const int k = b % K_;
  const int n = b / K_;
  const int tid = threadIdx.x;
  for (int e = tid; e < MID_ * V_; e += 256) {
    int c = e / V_, u = e % V_;
    size_t idx = ((size_t)n * KM_ + k * MID_ + c) * V_ + u;
    sx1[e] = x1[idx];
    sx2[e] = x2[idx];
  }
  __syncthreads();
  for (int e = tid; e < V_ * V_; e += 256) {        // g[u][v] = sum_c x1[c,u]*x2[c,v]
    int u = e / V_, v = e % V_;
    float s = 0.f;
    for (int c = 0; c < MID_; ++c) s = fmaf(sx1[c * V_ + u], sx2[c * V_ + v], s);
    gls[u * 26 + v] = s;
  }
  __syncthreads();
  if (tid < V_) {                                   // softmax over u (axis -2), * beta
    int v = tid;
    float m = -3.4e38f;
    for (int u = 0; u < V_; ++u) m = fmaxf(m, gls[u * 26 + v]);
    float s = 0.f;
    for (int u = 0; u < V_; ++u) s += __expf(gls[u * 26 + v] - m);
    float inv = beta[k] / s;
    for (int u = 0; u < V_; ++u) gls[u * 26 + v] = __expf(gls[u * 26 + v] - m) * inv;
  }
  __syncthreads();
  const float al = alpha[k];
  for (int c = 0; c < MID_; ++c) {
    __bf16* out = bagg + ((size_t)b * MID_ + c) * (VP_ * VP_);
    for (int e = tid; e < VP_ * VP_; e += 256) {
      int vr = e / VP_, uc = e % VP_;
      float val = 0.f;
      if (vr < V_ && uc < V_) {
        val = tanhf(sx1[c * V_ + uc] - sx2[c * V_ + vr]) * al
            + Am[((size_t)k * V_ + uc) * V_ + vr]
            + gls[uc * 26 + vr];
      }
      out[e] = (__bf16)val;
    }
  }
}

// ================= K3: pre GEMM  pre_x = relu(bn(pre_w @ x)) ==================
// block = 8 waves, N-tile 128 spatial, all 12 M-tiles; x panel in LDS via TDM.
// A-frags batched in register groups so loads issue as one clause -> one wait ->
// back-to-back WMMAs.
__global__ __launch_bounds__(256) void k3_pre_gemm(const __bf16* __restrict__ xbf,
                                                   const __bf16* __restrict__ pwbf,
                                                   const float* __restrict__ pre_b,
                                                   const float* __restrict__ pre_g,
                                                   const float* __restrict__ pre_be,
                                                   __bf16* __restrict__ pre_x) {
  __shared__ __align__(1024) __bf16 panel[C_ * 128];   // 64 KB: x[c][s0..s0+127]
  const int sg = blockIdx.x;          // 0..24
  const int n  = blockIdx.y;          // 0..31
  const int s0 = sg * 128;
  const int tid = threadIdx.x;
  const __bf16* xsrc = xbf + (size_t)n * C_ * SP_ + s0;
#if HAVE_TDM
  if (tid < 32) {  // one wave issues the DMA
    tdm_load_2d((unsigned)(uintptr_t)panel, xsrc,
                /*tile0=*/128, /*tile1=*/C_,
                /*td0=*/SP_, /*td1=*/C_, /*stride0=*/SP_);
    __builtin_amdgcn_s_wait_tensorcnt(0);
  }
#else
  {
    const uint4* src = (const uint4*)(xsrc + (size_t)tid * SP_);
    uint4* dst = (uint4*)(panel + tid * 128);
#pragma unroll
    for (int i = 0; i < 16; ++i) dst[i] = src[i];
  }
#endif
  __syncthreads();

  const int w = tid >> 5, l = tid & 31, hi = l >> 4, lm = l & 15;
  v8f acc[12];
#pragma unroll
  for (int m = 0; m < 12; ++m) acc[m] = vzero8();

  for (int kk = 0; kk < 8; ++kk) {
    const int c0 = kk * 32;
    v16bf bf;
#pragma unroll
    for (int i = 0; i < 16; ++i) bf[i] = panel[(c0 + (i < 8 ? i + 8 * hi : 8 + i + 8 * hi)) * 128 + 16 * w + lm];
    // prefetch next K-step of the weight panel into cache
    if (kk < 7)
      __builtin_prefetch(pwbf + (size_t)lm * 16 * C_ + c0 + 32, 0, 1);
#pragma unroll
    for (int mg = 0; mg < 2; ++mg) {
      v16bf af[6];
#pragma unroll
      for (int i = 0; i < 6; ++i)
        af[i] = load_afrag(pwbf + (size_t)((mg * 6 + i) * 16 + lm) * C_ + c0, hi);
#pragma unroll
      for (int i = 0; i < 6; ++i)
        acc[mg * 6 + i] = wmma_bf16(af[i], bf, acc[mg * 6 + i]);
    }
  }

  const int s = s0 + 16 * w + lm;
  const int t = s / V_, v = s % V_;
#pragma unroll
  for (int m = 0; m < 12; ++m) {
#pragma unroll
    for (int r = 0; r < 8; ++r) {
      int kc = m * 16 + r + hi * 8;
      float val = (acc[m][r] + pre_b[kc]) * (pre_g[kc] * RS_BN) + pre_be[kc];
      val = fmaxf(val, 0.0f);
      pre_x[(((size_t)n * KM_ + kc) * T_ + t) * VP_ + v] = (__bf16)val;
    }
  }
}

// ================= K4: graph aggregation  y[t,u] = sum_v pre[t,v]*Afull[u,v] ==
__global__ __launch_bounds__(256) void k4_agg(const __bf16* __restrict__ pre_x,
                                              const __bf16* __restrict__ bagg,
                                              __bf16* __restrict__ ybf) {
  const int b = blockIdx.x;           // n*KM_ + kc, 6144 blocks
  const int n = b / KM_, kc = b % KM_;
  const int k = kc / MID_, c = kc % MID_;
  const int tid = threadIdx.x, w = tid >> 5, l = tid & 31, hi = l >> 4, lm = l & 15;
  const int t0 = w * 16;

  const __bf16* arow = pre_x + (((size_t)n * KM_ + kc) * T_ + t0 + lm) * VP_;
  v16bf af = load_afrag(arow, hi);
  const __bf16* bbase = bagg + (((size_t)n * K_ + k) * MID_ + c) * (VP_ * VP_) + lm;
  v16bf b0, b1;
#pragma unroll
  for (int i = 0; i < 16; ++i) {
    int kr = (i < 8 ? i + 8 * hi : 8 + i + 8 * hi) * VP_;
    b0[i] = bbase[kr];
    b1[i] = bbase[kr + 16];
  }
  v8f acc0 = wmma_bf16(af, b0, vzero8());
  v8f acc1 = wmma_bf16(af, b1, vzero8());

  const size_t ybase = (size_t)kc * (N_ * SP_) + (size_t)n * SP_;
#pragma unroll
  for (int r = 0; r < 8; ++r) {
    int t = t0 + r + hi * 8;
    ybf[ybase + (size_t)t * V_ + lm] = (__bf16)acc0[r];
    int u = 16 + lm;
    if (u < V_) ybf[ybase + (size_t)t * V_ + u] = (__bf16)acc1[r];
  }
}

// ================= K5: post GEMM + bn + residual + relu =======================
__global__ __launch_bounds__(256) void k5_post_gemm(const __bf16* __restrict__ ybf,
                                                    const __bf16* __restrict__ qwbf,
                                                    const float* __restrict__ post_b,
                                                    const float* __restrict__ bn_g,
                                                    const float* __restrict__ bn_b,
                                                    const float* __restrict__ xres,
                                                    float* __restrict__ out) {
  __shared__ __align__(1024) __bf16 panel[KM_ * 128]; // 48 KB: y[kc][s0..s0+127]
  const int sg = blockIdx.x;          // 0..24
  const int n  = blockIdx.y;          // 0..31
  const int s0 = sg * 128;
  const int tid = threadIdx.x;
  const __bf16* ysrc = ybf + (size_t)n * SP_ + s0;    // row stride N_*SP_
#if HAVE_TDM
  if (tid < 32) {
    tdm_load_2d((unsigned)(uintptr_t)panel, ysrc,
                /*tile0=*/128, /*tile1=*/KM_,
                /*td0=*/(unsigned)(N_ * SP_), /*td1=*/KM_, /*stride0=*/(unsigned)(N_ * SP_));
    __builtin_amdgcn_s_wait_tensorcnt(0);
  }
#else
  if (tid < KM_) {
    const uint4* src = (const uint4*)(ysrc + (size_t)tid * (N_ * SP_));
    uint4* dst = (uint4*)(panel + tid * 128);
#pragma unroll
    for (int i = 0; i < 16; ++i) dst[i] = src[i];
  }
#endif
  __syncthreads();

  const int w = tid >> 5, l = tid & 31, hi = l >> 4, lm = l & 15;
  v8f acc[16];
#pragma unroll
  for (int m = 0; m < 16; ++m) acc[m] = vzero8();

  for (int kk = 0; kk < 6; ++kk) {
    const int c0 = kk * 32;
    v16bf bf;
#pragma unroll
    for (int i = 0; i < 16; ++i) bf[i] = panel[(c0 + (i < 8 ? i + 8 * hi : 8 + i + 8 * hi)) * 128 + 16 * w + lm];
    if (kk < 5)
      __builtin_prefetch(qwbf + (size_t)lm * 16 * KM_ + c0 + 32, 0, 1);
#pragma unroll
    for (int mg = 0; mg < 2; ++mg) {
      v16bf af[8];
#pragma unroll
      for (int i = 0; i < 8; ++i)
        af[i] = load_afrag(qwbf + (size_t)((mg * 8 + i) * 16 + lm) * KM_ + c0, hi);
#pragma unroll
      for (int i = 0; i < 8; ++i)
        acc[mg * 8 + i] = wmma_bf16(af[i], bf, acc[mg * 8 + i]);
    }
  }

  const int s = s0 + 16 * w + lm;     // flattened (t,v)
#pragma unroll
  for (int m = 0; m < 16; ++m) {
#pragma unroll
    for (int r = 0; r < 8; ++r) {
      int co = m * 16 + r + hi * 8;
      size_t idx = ((size_t)n * OUT_ + co) * SP_ + s;
      float val = (acc[m][r] + post_b[co]) * (bn_g[co] * RS_BN) + bn_b[co] + xres[idx];
      out[idx] = fmaxf(val, 0.0f);
    }
  }
}

// ================= host launcher =================
extern "C" void kernel_launch(void* const* d_in, const int* in_sizes, int n_in,
                              void* d_out, int out_size, void* d_ws, size_t ws_size,
                              hipStream_t stream) {
  (void)in_sizes; (void)n_in; (void)out_size; (void)ws_size;
  const float* x      = (const float*)d_in[0];
  const float* Am     = (const float*)d_in[1];
  const float* alpha  = (const float*)d_in[2];
  const float* beta   = (const float*)d_in[3];
  const float* pre_w  = (const float*)d_in[4];
  const float* pre_b  = (const float*)d_in[5];
  const float* pre_g  = (const float*)d_in[6];
  const float* pre_be = (const float*)d_in[7];
  const float* c1w    = (const float*)d_in[8];
  const float* c1b    = (const float*)d_in[9];
  const float* c2w    = (const float*)d_in[10];
  const float* c2b    = (const float*)d_in[11];
  const float* post_w = (const float*)d_in[12];
  const float* post_b = (const float*)d_in[13];
  const float* bn_g   = (const float*)d_in[14];
  const float* bn_b   = (const float*)d_in[15];
  float* out = (float*)d_out;

  char* ws = (char*)d_ws;
  __bf16* xbf   = (__bf16*)(ws + OFF_XBF);
  __bf16* prex  = (__bf16*)(ws + OFF_PREX);
  __bf16* ybf   = (__bf16*)(ws + OFF_Y);
  __bf16* bagg  = (__bf16*)(ws + OFF_BAGG);
  float*  tmp   = (float*)(ws + OFF_TMP);
  float*  x1    = (float*)(ws + OFF_X1);
  float*  x2    = (float*)(ws + OFF_X2);
  __bf16* pwbf  = (__bf16*)(ws + OFF_PW);
  __bf16* qwbf  = (__bf16*)(ws + OFF_QW);

  k0_mean_cvt<<<dim3(N_ * C_), 128, 0, stream>>>(x, xbf, tmp);
  kw_cvt<<<dim3((KM_ * C_) / 256), 256, 0, stream>>>(pre_w, post_w, pwbf, qwbf);
  k1_x1x2<<<dim3(N_ * KM_), 32, 0, stream>>>(tmp, c1w, c1b, c2w, c2b, x1, x2);
  k2_bagg<<<dim3(N_ * K_), 256, 0, stream>>>(x1, x2, Am, alpha, beta, bagg);
  // zero pre_x so the v=25..31 K-dim padding is clean (B rows there are zero too)
  (void)hipMemsetAsync(prex, 0, SZ_PREX, stream);
  k3_pre_gemm<<<dim3(SP_ / 128, N_), 256, 0, stream>>>(xbf, pwbf, pre_b, pre_g, pre_be, prex);
  k4_agg<<<dim3(N_ * KM_), 256, 0, stream>>>(prex, bagg, ybf);
  k5_post_gemm<<<dim3(SP_ / 128, N_), 256, 0, stream>>>(ybf, qwbf, post_b, bn_g, bn_b, x, out);
}
